// Reformer_34875134443675
// MI455X (gfx1250) — matrix operand order repaired and checked by reading, not compiled, not run on hardware
//
#include <hip/hip_runtime.h>
#include <hip/hip_bf16.h>
#include <cstdint>
#include <cstddef>

// ---------------------------------------------------------------------------
// Reformer forward for MI455X (gfx1250, wave32, WMMA + TDM).
// GEMM-class math -> v_wmma_f32_16x16x32_bf16 (f32 accumulate).
// Weight tiles (shared by all 8 waves of a block) -> tensor_load_to_lds.
// ---------------------------------------------------------------------------

typedef __attribute__((ext_vector_type(8)))  __bf16 v8bf;
typedef __attribute__((ext_vector_type(16))) __bf16 v16bf;
typedef __attribute__((ext_vector_type(8)))  float  v8f;

typedef __attribute__((ext_vector_type(4))) unsigned int rf_u32x4;
typedef __attribute__((ext_vector_type(8))) int          rf_i32x8;
typedef __attribute__((ext_vector_type(4))) int          rf_i32x4;

union Frag16 {
    v16bf v;
    v8bf  h[2];
};

#if defined(__has_builtin)
#  if __has_builtin(__builtin_amdgcn_tensor_load_to_lds)
#    define RF_HAS_TDM 1
#  endif
#endif
#ifndef RF_HAS_TDM
#  define RF_HAS_TDM 0
#endif

__device__ __forceinline__ float rf_gelu(float x) {
    return 0.5f * x * (1.0f + erff(x * 0.70710678118654752f));
}

#if RF_HAS_TDM
// 2-D tensor DMA: copy a [rows x k_elems] bf16 tile (row stride = stride_elems)
// from global memory into LDS at byte offset lds_off. D# per CDNA5 ISA §8.3/8.4.
__device__ __forceinline__ void rf_tdm_load_2d(const void* gptr,
                                               unsigned lds_off,
                                               unsigned k_elems,
                                               unsigned rows,
                                               unsigned stride_elems)
{
    const unsigned long long ga = (unsigned long long)(uintptr_t)gptr;

    rf_u32x4 g0;
    g0[0] = 1u;                                       // count=1, user descriptor
    g0[1] = lds_off;                                  // lds_addr (bytes)
    g0[2] = (unsigned)(ga & 0xffffffffull);           // global_addr[31:0]
    g0[3] = (unsigned)((ga >> 32) & 0x1ffffffull)     // global_addr[56:32]
          | (2u << 30);                               // type = 2 ("image")

    rf_i32x8 g1;
    g1[0] = (int)(1u << 16);                          // wg_mask=0, data_size=1 (2B)
    g1[1] = (int)((k_elems & 0xffffu) << 16);         // abar=0 | tensor_dim0[15:0]
    g1[2] = (int)(((k_elems >> 16) & 0xffffu)         // tensor_dim0[31:16]
          | ((rows & 0xffffu) << 16));                // tensor_dim1[15:0]
    g1[3] = (int)((k_elems & 0xffffu) << 16);         // tensor_dim1[31:16]=0 | tile_dim0
    g1[4] = (int)(rows & 0xffffu);                    // tile_dim1 | tile_dim2=0
    g1[5] = (int)stride_elems;                        // tensor_dim0_stride[31:0]
    g1[6] = 0;                                        // stride hi | dim1_stride lo
    g1[7] = 0;                                        // dim1_stride hi

    rf_i32x4 z4;
    z4[0] = 0; z4[1] = 0; z4[2] = 0; z4[3] = 0;

#if defined(__clang_major__) && (__clang_major__ >= 23)
    rf_i32x8 z8;
    for (int i = 0; i < 8; ++i) z8[i] = 0;
    __builtin_amdgcn_tensor_load_to_lds(g0, g1, z4, z4, z8, 0);
#else
    __builtin_amdgcn_tensor_load_to_lds(g0, g1, z4, z4, 0);
#endif
}
#endif // RF_HAS_TDM

// ---------------------------------------------------------------------------
// Generic bf16 WMMA GEMM: out[N,M] = A[N,K] @ Wt[M,K]^T (+bias, act, resid)
// 8 waves / block; each wave computes TWO adjacent 16x16 tiles sharing one
// A fragment; grid = (ceil(N/128), M/32). K must be a multiple of 128.
// use_tdm: stage the block's 32xK weight tile in LDS via the Tensor Data
// Mover (ds_load path); otherwise stream from global with chunk prefetch.
// ---------------------------------------------------------------------------
__global__ __launch_bounds__(256)
void rf_gemm_bf16(const __bf16* __restrict__ A, const __bf16* __restrict__ Wt,
                  const float* __restrict__ bias, const float* __restrict__ resid,
                  float* __restrict__ outF, __bf16* __restrict__ outB,
                  int N, int M, int K, int act, int use_tdm)
{
    extern __shared__ __bf16 sB[];                 // [32][K] when use_tdm

    const int lane  = threadIdx.x & 31;
    const int wave  = threadIdx.x >> 5;
    const int rbase = (blockIdx.x * 8 + wave) * 16;
    const int cbase = blockIdx.y * 32;
    const int ln    = lane & 15;
    const int hi    = lane >> 4;                   // half-wave select

    if (use_tdm) {
#if RF_HAS_TDM
        if (wave == 0) {
            rf_tdm_load_2d(Wt + (size_t)cbase * K, 0u,
                           (unsigned)K, 32u, (unsigned)K);
            __builtin_amdgcn_s_wait_tensorcnt(0);
        }
#else
        // Cooperative fallback: global -> LDS copy of the 32xK weight tile.
        for (int e = threadIdx.x * 8; e < 32 * K; e += 256 * 8) {
            const int r = e / K, c = e % K;
            *(v8bf*)(sB + e) = *(const v8bf*)(Wt + (size_t)(cbase + r) * K + c);
        }
#endif
        __syncthreads();
    }

    if (rbase < N) {
        // A fragment: lane<16 -> K {k0..k0+15}; lane>=16 -> +16
        const __bf16* arow = A + (size_t)(rbase + ln) * K + hi * 16;

        v8f acc0, acc1;
#pragma unroll
        for (int i = 0; i < 8; ++i) { acc0[i] = 0.0f; acc1[i] = 0.0f; }

        if (use_tdm) {
            // B fragments from LDS (address space stays LDS -> ds_load_b128)
            const __bf16* b0 = sB + (size_t)ln * K + hi * 16;
            const __bf16* b1 = b0 + (size_t)16 * K;
            for (int k0 = 0; k0 < K; k0 += 32) {
                Frag16 a, f0, f1;
                a.h[0] = *(const v8bf*)(arow + k0);
                a.h[1] = *(const v8bf*)(arow + k0 + 8);
                f0.v   = *(const v16bf*)(b0 + k0);
                f1.v   = *(const v16bf*)(b1 + k0);
                acc0 = __builtin_amdgcn_wmma_f32_16x16x32_bf16(
                    false, a.v, false, f0.v, (short)0, acc0, false, false);
                acc1 = __builtin_amdgcn_wmma_f32_16x16x32_bf16(
                    false, a.v, false, f1.v, (short)0, acc1, false, false);
            }
        } else {
            // Streaming path (pooling GEMM): global loads + chunk prefetch.
            const __bf16* b0 = Wt + (size_t)(cbase + ln) * K + hi * 16;
            const __bf16* b1 = b0 + (size_t)16 * K;
            for (int kc = 0; kc < K; kc += 128) {
                __builtin_prefetch(arow + kc + 4096, 0, 1);  // speculative
                __builtin_prefetch(b0 + kc + 4096, 0, 1);
#pragma unroll
                for (int ku = 0; ku < 128; ku += 32) {
                    const int k0 = kc + ku;
                    Frag16 a, f0, f1;
                    a.h[0] = *(const v8bf*)(arow + k0);
                    a.h[1] = *(const v8bf*)(arow + k0 + 8);
                    f0.v   = *(const v16bf*)(b0 + k0);
                    f1.v   = *(const v16bf*)(b1 + k0);
                    acc0 = __builtin_amdgcn_wmma_f32_16x16x32_bf16(
                        false, a.v, false, f0.v, (short)0, acc0, false, false);
                    acc1 = __builtin_amdgcn_wmma_f32_16x16x32_bf16(
                        false, a.v, false, f1.v, (short)0, acc1, false, false);
                }
            }
        }

        const int col0 = cbase + ln;
        const int col1 = cbase + 16 + ln;
#pragma unroll
        for (int r = 0; r < 8; ++r) {
            const int row = rbase + hi * 8 + r;
            float t0 = acc0[r];
            float t1 = acc1[r];
            if (bias)  { t0 += bias[col0];  t1 += bias[col1]; }
            if (act)   { t0 = rf_gelu(t0);  t1 = rf_gelu(t1); }
            if (resid) {
                t0 += resid[(size_t)row * M + col0];
                t1 += resid[(size_t)row * M + col1];
            }
            if (outF) {
                outF[(size_t)row * M + col0] = t0;
                outF[(size_t)row * M + col1] = t1;
            }
            if (outB) {
                outB[(size_t)row * M + col0] = (__bf16)t0;
                outB[(size_t)row * M + col1] = (__bf16)t1;
            }
        }
    }
}

// ---------------------------------------------------------------------------
// Embedding + axial positional embedding; writes both reversible streams.
// ---------------------------------------------------------------------------
__global__ __launch_bounds__(256)
void rf_embed(const int* __restrict__ tok, const float* __restrict__ emb,
              const float* __restrict__ pos1, const float* __restrict__ pos2,
              float* __restrict__ x1, float* __restrict__ x2)
{
    int idx = blockIdx.x * 256 + threadIdx.x;
    if (idx >= 64 * 384 * 128) return;
    const int d  = idx & 127;
    const int bl = idx >> 7;
    const int l  = bl % 384;
    const int t  = tok[bl];
    const float v = emb[(size_t)t * 128 + d]
                  + pos1[(l / 25) * 128 + d]
                  + pos2[(l % 25) * 128 + d];
    x1[idx] = v;
    x2[idx] = v;
}

// ---------------------------------------------------------------------------
// LayerNorm over D=128, one row per thread; optional (X+X2)*scale pre-combine.
// ---------------------------------------------------------------------------
__global__ __launch_bounds__(256)
void rf_ln(const float* __restrict__ X, const float* __restrict__ X2, float scale,
           const float* __restrict__ g, const float* __restrict__ bta,
           __bf16* __restrict__ out, int N)
{
    const int row = blockIdx.x * 256 + threadIdx.x;
    if (row >= N) return;
    const float* xr  = X + (size_t)row * 128;
    const float* x2r = X2 ? (X2 + (size_t)row * 128) : nullptr;

    float mu = 0.0f;
    for (int d = 0; d < 128; ++d) {
        float v = xr[d];
        if (x2r) v = (v + x2r[d]) * scale;
        mu += v;
    }
    mu *= (1.0f / 128.0f);

    float var = 0.0f;
    for (int d = 0; d < 128; ++d) {
        float v = xr[d];
        if (x2r) v = (v + x2r[d]) * scale;
        const float t = v - mu;
        var += t * t;
    }
    var *= (1.0f / 128.0f);
    const float rs = rsqrtf(var + 1e-5f);

    __bf16* orow = out + (size_t)row * 128;
    for (int d = 0; d < 128; ++d) {
        float v = xr[d];
        if (x2r) v = (v + x2r[d]) * scale;
        orow[d] = (__bf16)((v - mu) * rs * g[d] + bta[d]);
    }
}

// ---------------------------------------------------------------------------
// Per (b,h,l): Qs = qk * DH^-0.5 (bf16, [B,H,L,16]); Kh = l2-normalized qk
// (bf16, [B,H,L,16]); Vt = V transposed (bf16, [B,H,16,L]).
// ---------------------------------------------------------------------------
__global__ __launch_bounds__(256)
void rf_qkv_prep(const float* __restrict__ QK, const float* __restrict__ V,
                 __bf16* __restrict__ Qs, __bf16* __restrict__ Kh,
                 __bf16* __restrict__ Vt)
{
    int idx = blockIdx.x * 256 + threadIdx.x;   // (b*8+hd)*384 + l
    if (idx >= 64 * 8 * 384) return;
    const int l  = idx % 384;
    const int bh = idx / 384;
    const int hd = bh % 8;
    const int b  = bh / 8;

    const float* src = QK + ((size_t)b * 384 + l) * 128 + hd * 16;
    float q[16];
    float ss = 0.0f;
#pragma unroll
    for (int k = 0; k < 16; ++k) { q[k] = src[k]; ss += q[k] * q[k]; }
    const float inv = 1.0f / fmaxf(sqrtf(ss), 1e-12f);

    __bf16* qd = Qs + (size_t)idx * 16;
    __bf16* kd = Kh + (size_t)idx * 16;
#pragma unroll
    for (int k = 0; k < 16; ++k) {
        qd[k] = (__bf16)(q[k] * 0.25f);        // DH^-0.5 = 1/4 exactly
        kd[k] = (__bf16)(q[k] * inv);
    }

    const float* vs = V + ((size_t)b * 384 + l) * 128 + hd * 16;
#pragma unroll
    for (int k = 0; k < 16; ++k)
        Vt[((size_t)bh * 16 + k) * 384 + l] = (__bf16)vs[k];
}

// ---------------------------------------------------------------------------
// Attention: one wave per (qtile, head, batch). WMMA QK^T (K padded 16->32),
// masks in C-layout, LDS-staged row softmax, WMMA P@V against Vt.
// ---------------------------------------------------------------------------
__global__ __launch_bounds__(32)
void rf_attn(const __bf16* __restrict__ Qs, const __bf16* __restrict__ Kh,
             const __bf16* __restrict__ Vt, __bf16* __restrict__ O)
{
    __shared__ float  sd[16][388];   // stride 388 dwords -> bank-conflict-free
    __shared__ __bf16 sp[16][384];
    __shared__ float  srs[16];

    const int lane  = threadIdx.x;
    const int ln    = lane & 15;
    const int hi    = lane >> 4;
    const int qt    = blockIdx.x;
    const int hd    = blockIdx.y;
    const int bb    = blockIdx.z;
    const int qbase = qt * 16;
    const size_t headL = ((size_t)bb * 8 + hd) * 384;

    // Q fragment: valid K = 0..15, upper K zero-padded
    Frag16 aq;
#pragma unroll
    for (int i = 0; i < 16; ++i) aq.v[i] = (__bf16)0.0f;
    if (hi == 0) aq.v = *(const v16bf*)(Qs + (headL + qbase + ln) * 16);

    for (int jt = 0; jt < 24; ++jt) {
        const int jbase = jt * 16;
        Frag16 bk;
#pragma unroll
        for (int i = 0; i < 16; ++i) bk.v[i] = (__bf16)0.0f;
        if (hi == 0)     // lanes 0-15 carry K=0..15 (column = key jbase+ln)
            bk.v = *(const v16bf*)(Kh + (headL + jbase + ln) * 16);

        v8f c;
#pragma unroll
        for (int i = 0; i < 8; ++i) c[i] = 0.0f;
        c = __builtin_amdgcn_wmma_f32_16x16x32_bf16(
            false, aq.v, false, bk.v, (short)0, c, false, false);

#pragma unroll
        for (int r = 0; r < 8; ++r) {
            const int i = qbase + hi * 8 + r;
            const int j = jbase + ln;
            float d = c[r];
            if (j > i)       d = -1e9f;    // causal mask
            else if (j == i) d = -5e4f;    // token-self mask
            sd[hi * 8 + r][jbase + ln] = d;
        }
    }
    __syncthreads();

    if (lane < 16) {
        float m = -3.4e38f;
        for (int j = 0; j < 384; ++j) m = fmaxf(m, sd[lane][j]);
        float s = 0.0f;
        for (int j = 0; j < 384; ++j) {
            const float e = __expf(sd[lane][j] - m);
            s += e;
            sp[lane][j] = (__bf16)e;       // unnormalized; scale at output
        }
        srs[lane] = s;
    }
    __syncthreads();

    v8f oacc;
#pragma unroll
    for (int i = 0; i < 8; ++i) oacc[i] = 0.0f;
    const __bf16* sprow = &sp[ln][0];
    const __bf16* vbase = Vt + headL * 16 + (size_t)ln * 384;
    for (int kt = 0; kt < 12; ++kt) {
        const int k0 = kt * 32;
        Frag16 a, bv;
        a.h[0] = *(const v8bf*)(sprow + k0 + hi * 16);
        a.h[1] = *(const v8bf*)(sprow + k0 + 8 + hi * 16);
        bv.v   = *(const v16bf*)(vbase + k0 + hi * 16);
        oacc = __builtin_amdgcn_wmma_f32_16x16x32_bf16(
            false, a.v, false, bv.v, (short)0, oacc, false, false);
    }

#pragma unroll
    for (int r = 0; r < 8; ++r) {
        const int row = hi * 8 + r;
        const float inv = 1.0f / srs[row];
        O[((size_t)bb * 384 + qbase + row) * 128 + hd * 16 + ln] =
            (__bf16)(oacc[r] * inv);
    }
}

// ---------------------------------------------------------------------------
// fp32 [K,M] -> bf16 transposed [M,K] (weight prep each call).
// ---------------------------------------------------------------------------
__global__ __launch_bounds__(256)
void rf_twt(const float* __restrict__ src, __bf16* __restrict__ dst, int K, int M)
{
    const int idx = blockIdx.x * 256 + threadIdx.x;
    if (idx >= K * M) return;
    const int k = idx / M;
    const int m = idx % M;
    dst[(size_t)m * K + k] = (__bf16)src[idx];
}

// ---------------------------------------------------------------------------
// Host-side orchestration
// ---------------------------------------------------------------------------
extern "C" void kernel_launch(void* const* d_in, const int* in_sizes, int n_in,
                              void* d_out, int out_size, void* d_ws, size_t ws_size,
                              hipStream_t stream)
{
    (void)in_sizes; (void)n_in; (void)out_size; (void)ws_size;
    constexpr int B = 64, L = 384, D = 128, H = 8, DEPTH = 6, FF = 512, COUT = 128;
    constexpr int BL = B * L;

    const int*   x_enc = (const int*)  d_in[0];
    const float* emb   = (const float*)d_in[1];
    const float* pos1  = (const float*)d_in[2];
    const float* pos2  = (const float*)d_in[3];
    const float* ln1_g = (const float*)d_in[4];
    const float* ln1_b = (const float*)d_in[5];
    const float* Wqk   = (const float*)d_in[6];
    const float* Wv    = (const float*)d_in[7];
    const float* Wo    = (const float*)d_in[8];
    const float* bo    = (const float*)d_in[9];
    const float* ln2_g = (const float*)d_in[10];
    const float* ln2_b = (const float*)d_in[11];
    const float* W1    = (const float*)d_in[12];
    const float* b1    = (const float*)d_in[13];
    const float* W2    = (const float*)d_in[14];
    const float* b2    = (const float*)d_in[15];
    const float* lnf_g = (const float*)d_in[16];
    const float* lnf_b = (const float*)d_in[17];
    const float* Wp    = (const float*)d_in[18];
    const float* bp    = (const float*)d_in[19];

    char* ws = (char*)d_ws;
    size_t off = 0;
    auto alloc = [&](size_t bytes) -> void* {
        void* p = ws + off;
        off += (bytes + 255) & ~(size_t)255;
        return p;
    };

    float*  x1   = (float*) alloc((size_t)BL * D * 4);
    float*  x2   = (float*) alloc((size_t)BL * D * 4);
    float*  QK   = (float*) alloc((size_t)BL * D * 4);
    float*  Vf   = (float*) alloc((size_t)BL * D * 4);
    __bf16* hb   = (__bf16*)alloc((size_t)BL * D * 2);
    __bf16* Qs   = (__bf16*)alloc((size_t)BL * D * 2);
    __bf16* Khn  = (__bf16*)alloc((size_t)BL * D * 2);
    __bf16* Vt   = (__bf16*)alloc((size_t)BL * D * 2);
    __bf16* Ob   = (__bf16*)alloc((size_t)BL * D * 2);
    __bf16* gb   = (__bf16*)alloc((size_t)BL * D * 2);
    __bf16* f1   = (__bf16*)alloc((size_t)BL * FF * 2);
    __bf16* fin  = (__bf16*)alloc((size_t)BL * D * 2);
    __bf16* WqkT = (__bf16*)alloc((size_t)DEPTH * D * D * 2);
    __bf16* WvT  = (__bf16*)alloc((size_t)DEPTH * D * D * 2);
    __bf16* WoT  = (__bf16*)alloc((size_t)DEPTH * D * D * 2);
    __bf16* W1T  = (__bf16*)alloc((size_t)DEPTH * D * FF * 2);
    __bf16* W2T  = (__bf16*)alloc((size_t)DEPTH * FF * D * 2);
    __bf16* WpT  = (__bf16*)alloc((size_t)L * D * COUT * 2);

    auto cdiv = [](int a, int b) { return (a + b - 1) / b; };

    // ---- weight prep: fp32 -> bf16 transposed -----------------------------
    for (int ly = 0; ly < DEPTH; ++ly) {
        rf_twt<<<cdiv(D * D, 256), 256, 0, stream>>>(Wqk + (size_t)ly * D * D,
                                                     WqkT + (size_t)ly * D * D, D, D);
        rf_twt<<<cdiv(D * D, 256), 256, 0, stream>>>(Wv + (size_t)ly * D * D,
                                                     WvT + (size_t)ly * D * D, D, D);
        rf_twt<<<cdiv(D * D, 256), 256, 0, stream>>>(Wo + (size_t)ly * D * D,
                                                     WoT + (size_t)ly * D * D, D, D);
        rf_twt<<<cdiv(D * FF, 256), 256, 0, stream>>>(W1 + (size_t)ly * D * FF,
                                                      W1T + (size_t)ly * D * FF, D, FF);
        rf_twt<<<cdiv(FF * D, 256), 256, 0, stream>>>(W2 + (size_t)ly * FF * D,
                                                      W2T + (size_t)ly * FF * D, FF, D);
    }
    rf_twt<<<cdiv(L * D * COUT, 256), 256, 0, stream>>>(Wp, WpT, L * D, COUT);

    // ---- embedding + positions into both reversible streams ---------------
    rf_embed<<<cdiv(BL * D, 256), 256, 0, stream>>>(x_enc, emb, pos1, pos2, x1, x2);

    const size_t shD  = (size_t)32 * D * 2;   //  8 KB LDS weight tile (K=D)
    const size_t shFF = (size_t)32 * FF * 2;  // 32 KB LDS weight tile (K=FF)

    // ---- reversible blocks -------------------------------------------------
    for (int ly = 0; ly < DEPTH; ++ly) {
        // attn(x2): LN -> QK/V projections
        rf_ln<<<cdiv(BL, 256), 256, 0, stream>>>(x2, nullptr, 1.0f,
                                                 ln1_g + ly * D, ln1_b + ly * D, hb, BL);
        rf_gemm_bf16<<<dim3(BL / 128, D / 32), 256, shD, stream>>>(
            hb, WqkT + (size_t)ly * D * D, nullptr, nullptr, QK, nullptr, BL, D, D, 0, 1);
        rf_gemm_bf16<<<dim3(BL / 128, D / 32), 256, shD, stream>>>(
            hb, WvT + (size_t)ly * D * D, nullptr, nullptr, Vf, nullptr, BL, D, D, 0, 1);
        rf_qkv_prep<<<cdiv(B * H * L, 256), 256, 0, stream>>>(QK, Vf, Qs, Khn, Vt);
        rf_attn<<<dim3(L / 16, H, B), 32, 0, stream>>>(Qs, Khn, Vt, Ob);
        // y1 = x1 + O@Wo + bo   (in-place into x1)
        rf_gemm_bf16<<<dim3(BL / 128, D / 32), 256, shD, stream>>>(
            Ob, WoT + (size_t)ly * D * D, bo + ly * D, x1, x1, nullptr, BL, D, D, 0, 1);

        // ff(y1): LN -> W1+gelu -> W2 ; y2 = x2 + ff (in-place into x2)
        rf_ln<<<cdiv(BL, 256), 256, 0, stream>>>(x1, nullptr, 1.0f,
                                                 ln2_g + ly * D, ln2_b + ly * D, gb, BL);
        rf_gemm_bf16<<<dim3(BL / 128, FF / 32), 256, shD, stream>>>(
            gb, W1T + (size_t)ly * D * FF, b1 + ly * FF, nullptr, nullptr, f1, BL, FF, D, 1, 1);
        rf_gemm_bf16<<<dim3(BL / 128, D / 32), 256, shFF, stream>>>(
            f1, W2T + (size_t)ly * FF * D, b2 + ly * D, x2, x2, nullptr, BL, D, FF, 0, 1);
    }

    // ---- final LN of (x1+x2)/2 and pooling projection ----------------------
    rf_ln<<<cdiv(BL, 256), 256, 0, stream>>>(x1, x2, 0.5f, lnf_g, lnf_b, fin, BL);
    rf_gemm_bf16<<<dim3(cdiv(B, 128), COUT / 32), 256, 0, stream>>>(
        fin, WpT, bp, nullptr, (float*)d_out, nullptr, B, COUT, L * D, 0, 0);
}
